// fpn_classifier_graph_13331578487474
// MI455X (gfx1250) — matrix operand order repaired and checked by reading, not compiled, not run on hardware
//
#include <hip/hip_runtime.h>

// ---------------------------------------------------------------------------
// FPN classifier head for MI455X (gfx1250, wave32, WMMA).
// Dominant cost: GEMM1 (1024 x 12544 x 1024). Strategy: bf16 inputs with f32
// accumulate via v_wmma_f32_16x16x32_bf16; operands resident in 192MB L2.
// Wave tile 32x64 (2x4 WMMA tiles) -> 8 WMMAs per 32-deep K step; 2-wave
// blocks (64x64 block tile) for spread; K loop unrolled 2x for pipelining.
// ---------------------------------------------------------------------------

typedef __attribute__((ext_vector_type(16))) __bf16 v16bf;
typedef __attribute__((ext_vector_type(8)))  __bf16 v8bf;
typedef __attribute__((ext_vector_type(8)))  float  v8f;

#define RNUM 1024          // B*N rois
#define CCH  256
#define PP   7
#define KC1  (CCH*PP*PP)   // 12544
#define HID  1024
#define NCLS 81
#define NCLSP 128          // padded to multiple of 64
#define NBX  324
#define NBXP 384
#define TFO  256

static __device__ __forceinline__ __bf16 f2bf(float f) { return (__bf16)f; }

// ---------------- ROI align (bilinear, FPN level select) -> bf16 ------------
__global__ __launch_bounds__(256) void roi_align_bf16(
    const float* __restrict__ p2, const float* __restrict__ p3,
    const float* __restrict__ p4, const float* __restrict__ p5,
    const float* __restrict__ rois, __bf16* __restrict__ feat)
{
  size_t idx = (size_t)blockIdx.x * blockDim.x + threadIdx.x;
  const size_t total = (size_t)RNUM * KC1;
  if (idx >= total) return;
  int n = (int)(idx / KC1);
  int r = (int)(idx % KC1);
  int c  = r / (PP*PP);
  int pq = r % (PP*PP);
  int py = pq / PP, px = pq % PP;

  float rx1 = rois[n*4+0], ry1 = rois[n*4+1];
  float rx2 = rois[n*4+2], ry2 = rois[n*4+3];
  float area = (rx2-rx1)*(ry2-ry1);
  float lv = rintf(log2f(sqrtf(area) * (1.0f/224.0f))) + 4.0f;  // RNE matches jnp.round
  int lvl = (int)fminf(fmaxf(lv, 2.0f), 5.0f);

  const float* fm; int H;
  if      (lvl == 2) { fm = p2; H = 256; }
  else if (lvl == 3) { fm = p3; H = 128; }
  else if (lvl == 4) { fm = p4; H = 64;  }
  else               { fm = p5; H = 32;  }
  // NOTE: reference uses feats[i][0] -> always batch 0's feature map.

  float by1 = ry1 * (1.0f/1024.0f), bx1 = rx1 * (1.0f/1024.0f);
  float by2 = ry2 * (1.0f/1024.0f), bx2 = rx2 * (1.0f/1024.0f);
  float gy = (float)py * (1.0f/6.0f), gx = (float)px * (1.0f/6.0f);
  float ys = (by1 + gy*(by2-by1)) * (float)(H-1);
  float xs = (bx1 + gx*(bx2-bx1)) * (float)(H-1);
  float y0f = floorf(ys), x0f = floorf(xs);
  float ly = ys - y0f, lx = xs - x0f;
  float hm1 = (float)(H-1);
  int y0  = (int)fminf(fmaxf(y0f,       0.f), hm1);
  int y1i = (int)fminf(fmaxf(y0f + 1.f, 0.f), hm1);
  int x0  = (int)fminf(fmaxf(x0f,       0.f), hm1);
  int x1i = (int)fminf(fmaxf(x0f + 1.f, 0.f), hm1);

  const float* base = fm + (size_t)c * H * H;
  float v00 = base[y0 *H + x0], v01 = base[y0 *H + x1i];
  float v10 = base[y1i*H + x0], v11 = base[y1i*H + x1i];
  float val = v00*(1.f-ly)*(1.f-lx) + v01*(1.f-ly)*lx
            + v10*ly*(1.f-lx)       + v11*ly*lx;
  bool valid = (ys >= 0.f) && (ys <= hm1) && (xs >= 0.f) && (xs <= hm1);
  feat[idx] = f2bf(valid ? val : 0.0f);
}

// ---------------- f32 -> bf16 with zero row padding -------------------------
__global__ __launch_bounds__(256) void cvt_pad_bf16(
    const float* __restrict__ src, __bf16* __restrict__ dst,
    int rows, int cols, int rowsPad)
{
  size_t total = (size_t)rowsPad * cols;
  for (size_t i = (size_t)blockIdx.x*blockDim.x + threadIdx.x; i < total;
       i += (size_t)gridDim.x * blockDim.x) {
    int rr = (int)(i / cols);
    dst[i] = (rr < rows) ? f2bf(src[i]) : f2bf(0.0f);
  }
}

// ---------------- Generic bf16 WMMA GEMM:  D = A[M,K] * W[N,K]^T ------------
// Block: 64 threads = 2 waves stacked in M; block tile 64x64.
// Wave tile 32x64 = 2 (M) x 4 (N) WMMA tiles -> 8 v_wmma per 32-deep K step.
// Requires M % 64 == 0, N % 64 == 0, K % 64 == 0 (padded buffers).
// mode: 0 = raw, 1 = +bias, 2 = +bias then leaky(0.01)
__global__ __launch_bounds__(64) void gemm_bf16_wmma(
    const __bf16* __restrict__ A, int lda,
    const __bf16* __restrict__ W, int ldw,
    const float* __restrict__ bias,
    float* __restrict__ outF, int ldcF,
    __bf16* __restrict__ outB, int ldcB,
    int K, int m_valid, int n_valid, int mode)
{
  const int lane = threadIdx.x & 31;
  const int wave = threadIdx.x >> 5;
  const int wm = blockIdx.x * 64 + wave * 32;
  const int wn = blockIdx.y * 64;
  const int row  = lane & 15;
  const int half = lane >> 4;
  const int kbA = half * 8;    // A: lanes 0-15 hold K{0-7,16-23}, 16-31 K{8-15,24-31}
  const int koB = half * 8;    // B: lanes 0-15 hold K{0-7,16-23}, 16-31 K{8-15,24-31}

  const __bf16* Ap[2];
  Ap[0] = A + (size_t)(wm + row) * lda;
  Ap[1] = Ap[0] + (size_t)16 * lda;
  const __bf16* Wp[4];
#pragma unroll
  for (int ni = 0; ni < 4; ++ni)
    Wp[ni] = W + (size_t)(wn + ni*16 + row) * ldw;   // row of W == column of B

  v8f acc[2][4] = {};
#pragma unroll 2
  for (int k = 0; k < K; k += 32) {
    v16bf af[2], bf_[4];
#pragma unroll
    for (int mi = 0; mi < 2; ++mi) {
      const __bf16* p = Ap[mi] + k;
      v8bf lo = *(const v8bf*)(p + kbA);
      v8bf hi = *(const v8bf*)(p + 16 + kbA);
#pragma unroll
      for (int e = 0; e < 8; ++e) { af[mi][e] = lo[e]; af[mi][8+e] = hi[e]; }
    }
#pragma unroll
    for (int ni = 0; ni < 4; ++ni) {
      const __bf16* p = Wp[ni] + k + koB;
      v8bf lo = *(const v8bf*)(p);
      v8bf hi = *(const v8bf*)(p + 16);
#pragma unroll
      for (int e = 0; e < 8; ++e) { bf_[ni][e] = lo[e]; bf_[ni][8+e] = hi[e]; }
    }
    if (k + 256 < K) {
      __builtin_prefetch(Ap[0] + k + 256, 0, 3);   // global_prefetch_b8
      __builtin_prefetch(Wp[0] + k + 256, 0, 3);
      __builtin_prefetch(Wp[2] + k + 256, 0, 3);
    }
#pragma unroll
    for (int mi = 0; mi < 2; ++mi)
#pragma unroll
      for (int ni = 0; ni < 4; ++ni)
        acc[mi][ni] = __builtin_amdgcn_wmma_f32_16x16x32_bf16(
            false, af[mi], false, bf_[ni], (short)0, acc[mi][ni], false, false);
  }

  // Epilogue. C/D layout: element e -> row half*8+e, col lane&15.
  const int cn = row;
  const int rb = half * 8;
#pragma unroll
  for (int mi = 0; mi < 2; ++mi) {
#pragma unroll
    for (int ni = 0; ni < 4; ++ni) {
      int gn = wn + ni*16 + cn;
      float bv = (mode != 0 && gn < n_valid) ? bias[gn] : 0.0f;
#pragma unroll
      for (int e = 0; e < 8; ++e) {
        int gm = wm + mi*16 + rb + e;
        float v = acc[mi][ni][e] + bv;
        if (mode == 2) v = (v < 0.0f) ? 0.01f * v : v;
        if (gm < m_valid && gn < n_valid) {
          if (outF) outF[(size_t)gm * ldcF + gn] = v;
          if (outB) outB[(size_t)gm * ldcB + gn] = f2bf(v);
        }
      }
    }
  }
}

// ---------------- BN (over rows) + ReLU -> bf16 -----------------------------
__global__ __launch_bounds__(256) void bn_relu_to_bf16(
    const float* __restrict__ Hm, const float* __restrict__ g,
    const float* __restrict__ b, __bf16* __restrict__ out, int M, int Nc)
{
  int j = blockIdx.x;
  __shared__ float s1[256], s2[256];
  float sum = 0.f, sq = 0.f;
  for (int i = threadIdx.x; i < M; i += 256) {
    float v = Hm[(size_t)i*Nc + j];
    sum += v; sq += v*v;
  }
  s1[threadIdx.x] = sum; s2[threadIdx.x] = sq;
  __syncthreads();
  for (int s = 128; s > 0; s >>= 1) {
    if (threadIdx.x < s) { s1[threadIdx.x] += s1[threadIdx.x+s];
                           s2[threadIdx.x] += s2[threadIdx.x+s]; }
    __syncthreads();
  }
  float mean = s1[0] / (float)M;
  float var  = s2[0] / (float)M - mean*mean;     // population var == h.var(0)
  float sc = rsqrtf(var + 0.001f) * g[j];
  float off = b[j];
  for (int i = threadIdx.x; i < M; i += 256) {
    float v = (Hm[(size_t)i*Nc + j] - mean) * sc + off;
    v = fmaxf(v, 0.0f);
    out[(size_t)i*Nc + j] = f2bf(v);
  }
}

// ---------------- row softmax (nc <= 128) -----------------------------------
__global__ __launch_bounds__(128) void softmax_rows(
    const float* __restrict__ logits, float* __restrict__ probs, int nc)
{
  int r = blockIdx.x, t = threadIdx.x;
  __shared__ float sm[128];
  float v = (t < nc) ? logits[(size_t)r*nc + t] : -3.4e38f;
  sm[t] = v; __syncthreads();
  for (int s = 64; s > 0; s >>= 1) { if (t < s) sm[t] = fmaxf(sm[t], sm[t+s]); __syncthreads(); }
  float m = sm[0]; __syncthreads();
  float e = (t < nc) ? expf(v - m) : 0.0f;
  sm[t] = e; __syncthreads();
  for (int s = 64; s > 0; s >>= 1) { if (t < s) sm[t] += sm[t+s]; __syncthreads(); }
  if (t < nc) probs[(size_t)r*nc + t] = e / sm[0];
}

// ---------------------------------------------------------------------------
extern "C" void kernel_launch(void* const* d_in, const int* in_sizes, int n_in,
                              void* d_out, int out_size, void* d_ws, size_t ws_size,
                              hipStream_t stream)
{
  (void)in_sizes; (void)n_in; (void)out_size; (void)ws_size;
  const float* p2      = (const float*)d_in[0];
  const float* p3      = (const float*)d_in[1];
  const float* p4      = (const float*)d_in[2];
  const float* p5      = (const float*)d_in[3];
  const float* rois    = (const float*)d_in[4];
  const float* conv1_w = (const float*)d_in[5];
  const float* conv1_b = (const float*)d_in[6];
  const float* bn1_g   = (const float*)d_in[7];
  const float* bn1_b   = (const float*)d_in[8];
  const float* conv2_w = (const float*)d_in[9];
  const float* conv2_b = (const float*)d_in[10];
  const float* bn2_g   = (const float*)d_in[11];
  const float* bn2_b   = (const float*)d_in[12];
  const float* cls_w   = (const float*)d_in[13];
  const float* cls_b   = (const float*)d_in[14];
  const float* box_w   = (const float*)d_in[15];
  const float* box_b   = (const float*)d_in[16];
  const float* tf1_w   = (const float*)d_in[17];
  const float* tf1_b   = (const float*)d_in[18];
  const float* tf2_w   = (const float*)d_in[19];
  const float* tf2_b   = (const float*)d_in[20];

  char* ws = (char*)d_ws;
  size_t off = 0;
  auto alloc = [&](size_t bytes) -> char* {
    char* p = ws + off; off += (bytes + 255) & ~(size_t)255; return p;
  };
  __bf16* featA = (__bf16*)alloc((size_t)RNUM  * KC1 * 2);
  __bf16* w1b   = (__bf16*)alloc((size_t)HID   * KC1 * 2);
  __bf16* w2b   = (__bf16*)alloc((size_t)HID   * HID * 2);
  __bf16* clsbw = (__bf16*)alloc((size_t)NCLSP * HID * 2);
  __bf16* boxbw = (__bf16*)alloc((size_t)NBXP  * HID * 2);
  __bf16* tf1bw = (__bf16*)alloc((size_t)HID   * HID * 2);
  __bf16* tf2bw = (__bf16*)alloc((size_t)TFO   * HID * 2);
  float*  H1    = (float*) alloc((size_t)RNUM  * HID * 4);
  __bf16* H1b   = (__bf16*)alloc((size_t)RNUM  * HID * 2);
  float*  H2    = (float*) alloc((size_t)RNUM  * HID * 4);
  __bf16* Sb    = (__bf16*)alloc((size_t)RNUM  * HID * 2);
  __bf16* mw1b  = (__bf16*)alloc((size_t)NCLSP * HID * 2);

  float* out        = (float*)d_out;
  float* out_logits = out;                                  // 1024*81
  float* out_probs  = out_logits + (size_t)RNUM * NCLS;     // 1024*81
  float* out_bbox   = out_probs  + (size_t)RNUM * NCLS;     // 1024*324
  float* out_mw     = out_bbox   + (size_t)RNUM * NBX;      // 81*256

  // Weight conversions (padded rows zero-filled).
  {
    size_t t;
    t = (size_t)HID*KC1;   cvt_pad_bf16<<<(t+255)/256,256,0,stream>>>(conv1_w, w1b,   HID,  KC1, HID);
    t = (size_t)HID*HID;   cvt_pad_bf16<<<(t+255)/256,256,0,stream>>>(conv2_w, w2b,   HID,  HID, HID);
    t = (size_t)NCLSP*HID; cvt_pad_bf16<<<(t+255)/256,256,0,stream>>>(cls_w,   clsbw, NCLS, HID, NCLSP);
    t = (size_t)NBXP*HID;  cvt_pad_bf16<<<(t+255)/256,256,0,stream>>>(box_w,   boxbw, NBX,  HID, NBXP);
    t = (size_t)HID*HID;   cvt_pad_bf16<<<(t+255)/256,256,0,stream>>>(tf1_w,   tf1bw, HID,  HID, HID);
    t = (size_t)TFO*HID;   cvt_pad_bf16<<<(t+255)/256,256,0,stream>>>(tf2_w,   tf2bw, TFO,  HID, TFO);
  }
  // ROI align -> bf16 feature matrix A [1024 x 12544]
  {
    size_t t = (size_t)RNUM * KC1;
    roi_align_bf16<<<(t+255)/256,256,0,stream>>>(p2, p3, p4, p5, rois, featA);
  }
  dim3 blk(64);
  // GEMM1 + bias -> H1 f32
  gemm_bf16_wmma<<<dim3(RNUM/64, HID/64), blk, 0, stream>>>(
      featA, KC1, w1b, KC1, conv1_b, H1, HID, nullptr, 0, KC1, RNUM, HID, 1);
  bn_relu_to_bf16<<<HID, 256, 0, stream>>>(H1, bn1_g, bn1_b, H1b, RNUM, HID);
  // GEMM2 + bias -> H2 f32
  gemm_bf16_wmma<<<dim3(RNUM/64, HID/64), blk, 0, stream>>>(
      H1b, HID, w2b, HID, conv2_b, H2, HID, nullptr, 0, HID, RNUM, HID, 1);
  bn_relu_to_bf16<<<HID, 256, 0, stream>>>(H2, bn2_g, bn2_b, Sb, RNUM, HID);
  // logits = shared @ cls_w^T + cls_b   (N padded 81->128)
  gemm_bf16_wmma<<<dim3(RNUM/64, NCLSP/64), blk, 0, stream>>>(
      Sb, HID, clsbw, HID, cls_b, out_logits, NCLS, nullptr, 0, HID, RNUM, NCLS, 1);
  softmax_rows<<<RNUM, 128, 0, stream>>>(out_logits, out_probs, NCLS);
  // bbox = shared @ box_w^T + box_b     (N padded 324->384)
  gemm_bf16_wmma<<<dim3(RNUM/64, NBXP/64), blk, 0, stream>>>(
      Sb, HID, boxbw, HID, box_b, out_bbox, NBX, nullptr, 0, HID, RNUM, NBX, 1);
  // mw1 = leaky(cls_w @ tf1_w^T + tf1_b) -> bf16 (M padded 81->128; pad rows
  // are A==0 so deterministic, and never read by valid mw2 output rows)
  gemm_bf16_wmma<<<dim3(NCLSP/64, HID/64), blk, 0, stream>>>(
      clsbw, HID, tf1bw, HID, tf1_b, nullptr, 0, mw1b, HID, HID, NCLSP, HID, 2);
  // mw = leaky(mw1 @ tf2_w^T + tf2_b) -> out (81 x 256)
  gemm_bf16_wmma<<<dim3(NCLSP/64, TFO/64), blk, 0, stream>>>(
      mw1b, HID, tf2bw, HID, tf2_b, out_mw, TFO, nullptr, 0, HID, NCLS, TFO, 2);
}
